// MultiHeadAttention_29171417874795
// MI455X (gfx1250) — matrix operand order repaired and checked by reading, not compiled, not run on hardware
//
#include <hip/hip_runtime.h>
#include <hip/hip_bf16.h>
#include <math.h>

#define B_DIM 4
#define S_LEN 2048
#define DMODEL 1024
#define NHEAD 16
#define DK 64

typedef __bf16 bf16_t;
typedef bf16_t v16bf __attribute__((ext_vector_type(16)));
typedef bf16_t v8bf  __attribute__((ext_vector_type(8)));
typedef float  v8f   __attribute__((ext_vector_type(8)));

__device__ __forceinline__ v8f wmma_bf16(v16bf a, v16bf b, v8f c) {
  return __builtin_amdgcn_wmma_f32_16x16x32_bf16(false, a, false, b, (short)0, c,
                                                 false, false);
}

// ---- Fragment loaders (per CDNA5 ISA 7.12.2 layouts) ----
// A (16x32, M=rows, K=cols): lane<16: row=lane, K={0..7,16..23};
//                            lane>=16: row=lane-16, K={8..15,24..31}
__device__ __forceinline__ v16bf load_a_bf16(const bf16_t* p, int ld, int lane) {
  int row = lane & 15;
  int kb = (lane < 16) ? 0 : 8;
  const bf16_t* r0 = p + (size_t)row * ld + kb;
  v8bf lo = *(const v8bf*)(r0);
  v8bf hi = *(const v8bf*)(r0 + 16);
  v16bf out;
#pragma unroll
  for (int j = 0; j < 8; ++j) { out[j] = lo[j]; out[8 + j] = hi[j]; }
  return out;
}

// A fragment from f32 source (used for LDS-resident softmax weights)
__device__ __forceinline__ v16bf load_a_f32(const float* p, int ld, int lane) {
  int row = lane & 15;
  int kb = (lane < 16) ? 0 : 8;
  const float* r0 = p + (size_t)row * ld + kb;
  v16bf out;
#pragma unroll
  for (int j = 0; j < 8; ++j) out[j] = (bf16_t)r0[j];
#pragma unroll
  for (int j = 0; j < 8; ++j) out[8 + j] = (bf16_t)r0[16 + j];
  return out;
}

// B (32x16, K=rows, N=cols) from row-major source M[N][K] (B = M^T):
// lane<16: n=lane, K={0..7,16..23}; lane>=16: n=lane-16, K={8..15,24..31}
__device__ __forceinline__ v16bf load_b_bf16(const bf16_t* p, int ld, int lane) {
  int row = lane & 15;
  int kb = (lane < 16) ? 0 : 8;
  const bf16_t* r0 = p + (size_t)row * ld + kb;
  v8bf lo = *(const v8bf*)(r0);
  v8bf hi = *(const v8bf*)(r0 + 16);
  v16bf out;
#pragma unroll
  for (int j = 0; j < 8; ++j) { out[j] = lo[j]; out[8 + j] = hi[j]; }
  return out;
}

// ---------------- f32 -> bf16 streaming convert ----------------
__global__ __launch_bounds__(256) void cvt_kernel(const float* __restrict__ src,
                                                  bf16_t* __restrict__ dst,
                                                  int n) {
  int i = (blockIdx.x * 256 + threadIdx.x) * 8;
  if (i < n) {
    const float4* s = (const float4*)(src + i);
    float4 a = s[0], b = s[1];
    v8bf o;
    o[0] = (bf16_t)a.x; o[1] = (bf16_t)a.y; o[2] = (bf16_t)a.z; o[3] = (bf16_t)a.w;
    o[4] = (bf16_t)b.x; o[5] = (bf16_t)b.y; o[6] = (bf16_t)b.z; o[7] = (bf16_t)b.w;
    *(v8bf*)(dst + i) = o;
  }
}

// ---------------- Projection kernel (Q, K): out[b,h,s,dk] ----------------
// Block = one 16-row s-tile; wave w computes 16x64 strip. Software-pipelined.
__global__ __launch_bounds__(512) void proj_kernel(
    const bf16_t* __restrict__ X, const bf16_t* __restrict__ W,
    const float* __restrict__ bias, bf16_t* __restrict__ out) {
  int lane = threadIdx.x & 31;
  int wave = threadIdx.x >> 5;
  int stg = blockIdx.x;  // global s-tile 0..511
  int o0 = wave * 64;
  const bf16_t* Xp = X + (size_t)stg * 16 * DMODEL;
  const bf16_t* W0 = W + (size_t)(o0 + 0) * DMODEL;
  const bf16_t* W1 = W + (size_t)(o0 + 16) * DMODEL;
  const bf16_t* W2 = W + (size_t)(o0 + 32) * DMODEL;
  const bf16_t* W3 = W + (size_t)(o0 + 48) * DMODEL;
  v8f acc0 = {}, acc1 = {}, acc2 = {}, acc3 = {};
  v16bf a = load_a_bf16(Xp, DMODEL, lane);
  v16bf b0 = load_b_bf16(W0, DMODEL, lane);
  v16bf b1 = load_b_bf16(W1, DMODEL, lane);
  v16bf b2 = load_b_bf16(W2, DMODEL, lane);
  v16bf b3 = load_b_bf16(W3, DMODEL, lane);
  for (int d0 = 0; d0 < DMODEL; d0 += 32) {
    v16bf an = a, bn0 = b0, bn1 = b1, bn2 = b2, bn3 = b3;
    int dn = d0 + 32;
    if (dn < DMODEL) {
      an = load_a_bf16(Xp + dn, DMODEL, lane);
      bn0 = load_b_bf16(W0 + dn, DMODEL, lane);
      bn1 = load_b_bf16(W1 + dn, DMODEL, lane);
      bn2 = load_b_bf16(W2 + dn, DMODEL, lane);
      bn3 = load_b_bf16(W3 + dn, DMODEL, lane);
    }
    acc0 = wmma_bf16(a, b0, acc0);
    acc1 = wmma_bf16(a, b1, acc1);
    acc2 = wmma_bf16(a, b2, acc2);
    acc3 = wmma_bf16(a, b3, acc3);
    a = an; b0 = bn0; b1 = bn1; b2 = bn2; b3 = bn3;
  }
  int b_idx = stg >> 7;
  int srow0 = (stg & 127) * 16;
  int h = wave;
  int n = lane & 15;
  bf16_t* op = out + (((size_t)(b_idx * NHEAD + h)) * S_LEN + srow0) * DK + n;
  v8f* accs[4] = {&acc0, &acc1, &acc2, &acc3};
#pragma unroll
  for (int i = 0; i < 4; ++i) {
    float bb = bias[o0 + i * 16 + n];
#pragma unroll
    for (int r = 0; r < 8; ++r) {
      int m = r + ((lane >> 4) << 3);
      op[(size_t)m * DK + i * 16] = (bf16_t)((*accs[i])[r] + bb);
    }
  }
}

// ---------------- V projection, transposed output: out[b,h,dk,s] ----------
__global__ __launch_bounds__(512) void projT_kernel(
    const bf16_t* __restrict__ X, const bf16_t* __restrict__ W,
    const float* __restrict__ bias, bf16_t* __restrict__ out) {
  int lane = threadIdx.x & 31;
  int wave = threadIdx.x >> 5;
  int bx = blockIdx.x;  // 512 blocks
  int b_idx = bx >> 7;
  int rem = bx & 127;
  int og = rem >> 5;  // 0..3
  int sg = rem & 31;  // 0..31
  int o0 = (og * 16 + wave) * 16;
  int s0 = sg * 64;
  const bf16_t* Wp = W + (size_t)o0 * DMODEL;
  const bf16_t* X0 = X + ((size_t)b_idx * S_LEN + s0 + 0) * DMODEL;
  const bf16_t* X1 = X + ((size_t)b_idx * S_LEN + s0 + 16) * DMODEL;
  const bf16_t* X2 = X + ((size_t)b_idx * S_LEN + s0 + 32) * DMODEL;
  const bf16_t* X3 = X + ((size_t)b_idx * S_LEN + s0 + 48) * DMODEL;
  v8f acc0 = {}, acc1 = {}, acc2 = {}, acc3 = {};
  v16bf a = load_a_bf16(Wp, DMODEL, lane);
  v16bf b0 = load_b_bf16(X0, DMODEL, lane);
  v16bf b1 = load_b_bf16(X1, DMODEL, lane);
  v16bf b2 = load_b_bf16(X2, DMODEL, lane);
  v16bf b3 = load_b_bf16(X3, DMODEL, lane);
  for (int d0 = 0; d0 < DMODEL; d0 += 32) {
    v16bf an = a, bn0 = b0, bn1 = b1, bn2 = b2, bn3 = b3;
    int dn = d0 + 32;
    if (dn < DMODEL) {
      an = load_a_bf16(Wp + dn, DMODEL, lane);
      bn0 = load_b_bf16(X0 + dn, DMODEL, lane);
      bn1 = load_b_bf16(X1 + dn, DMODEL, lane);
      bn2 = load_b_bf16(X2 + dn, DMODEL, lane);
      bn3 = load_b_bf16(X3 + dn, DMODEL, lane);
    }
    acc0 = wmma_bf16(a, b0, acc0);
    acc1 = wmma_bf16(a, b1, acc1);
    acc2 = wmma_bf16(a, b2, acc2);
    acc3 = wmma_bf16(a, b3, acc3);
    a = an; b0 = bn0; b1 = bn1; b2 = bn2; b3 = bn3;
  }
  int h = o0 >> 6, dk0 = o0 & 63;
  int n = lane & 15;
  bf16_t* op =
      out + (((size_t)(b_idx * NHEAD + h)) * DK + dk0) * S_LEN + s0 + n;
  v8f* accs[4] = {&acc0, &acc1, &acc2, &acc3};
#pragma unroll
  for (int r = 0; r < 8; ++r) {
    int m = r + ((lane >> 4) << 3);
    float bb = bias[o0 + m];
#pragma unroll
    for (int i = 0; i < 4; ++i)
      op[(size_t)m * S_LEN + i * 16] = (bf16_t)((*accs[i])[r] + bb);
  }
}

// ---------------- Fused attention kernel ----------------
// Block = 16 waves = 16 heads; 32-row q-tile per block.
__global__ __launch_bounds__(512) void attn_kernel(
    const bf16_t* __restrict__ Qp, const bf16_t* __restrict__ Kp,
    const bf16_t* __restrict__ VpT, bf16_t* __restrict__ val) {
  __shared__ float sW[NHEAD * 32 * 33];  // 67.6 KB, stride 33 kills conflicts
  int lane = threadIdx.x & 31;
  int h = threadIdx.x >> 5;
  int qt = blockIdx.x;  // 0..255
  int b = qt >> 6;
  int q0 = (qt & 63) * 32;
  const bf16_t* Qh = Qp + (((size_t)(b * NHEAD + h)) * S_LEN + q0) * DK;
  const bf16_t* Kh = Kp + ((size_t)(b * NHEAD + h)) * S_LEN * DK;
  const bf16_t* Vh = VpT + ((size_t)(b * NHEAD + h)) * DK * S_LEN;

  v16bf qa[2][2];
#pragma unroll
  for (int t = 0; t < 2; ++t) {
    qa[t][0] = load_a_bf16(Qh + (size_t)t * 16 * DK, DK, lane);
    qa[t][1] = load_a_bf16(Qh + (size_t)t * 16 * DK + 32, DK, lane);
  }
  v8f acc[2][4] = {};
  float* myS = sW + h * (32 * 33);
  const float inv_sqrt_dk = 0.125f;

  for (int k0 = 0; k0 < S_LEN; k0 += 32) {
    // prefetch next iteration's K rows and V row-chunks (global_prefetch_b8)
    if (k0 + 32 < S_LEN) {
      __builtin_prefetch(Kh + (size_t)(k0 + 32 + lane) * DK, 0, 3);
      __builtin_prefetch(Vh + (size_t)lane * S_LEN + k0 + 32, 0, 3);
      __builtin_prefetch(Vh + (size_t)(lane + 32) * S_LEN + k0 + 32, 0, 3);
    }
    // ---- scores S_h[32 x 32] = Q_h (32x64) x K_h^T, K frags reused ----
#pragma unroll
    for (int nt = 0; nt < 2; ++nt) {
      v16bf kb0 = load_b_bf16(Kh + (size_t)(k0 + nt * 16) * DK, DK, lane);
      v16bf kb1 = load_b_bf16(Kh + (size_t)(k0 + nt * 16) * DK + 32, DK, lane);
#pragma unroll
      for (int t = 0; t < 2; ++t) {
        v8f s = {};
        s = wmma_bf16(qa[t][0], kb0, s);
        s = wmma_bf16(qa[t][1], kb1, s);
        int n = (lane & 15) + nt * 16;
#pragma unroll
        for (int r = 0; r < 8; ++r) {
          int m = t * 16 + r + ((lane >> 4) << 3);
          myS[m * 33 + n] = s[r] * inv_sqrt_dk;
        }
      }
    }
    __syncthreads();
    // ---- double softmax over heads; each thread owns 2 (q,k) elements ----
#pragma unroll
    for (int e = 0; e < 2; ++e) {
      int idx = threadIdx.x + e * 512;
      int m = idx >> 5;
      int n = idx & 31;
      float v[NHEAD];
      float mx = -3.4e38f;
#pragma unroll
      for (int j = 0; j < NHEAD; ++j) {
        v[j] = sW[j * (32 * 33) + m * 33 + n];
        mx = fmaxf(mx, v[j]);
      }
      float sum = 0.f;
#pragma unroll
      for (int j = 0; j < NHEAD; ++j) { v[j] = __expf(v[j] - mx); sum += v[j]; }
      float inv = 1.f / sum;
      float mx2 = -3.4e38f;
#pragma unroll
      for (int j = 0; j < NHEAD; ++j) {
        v[j] = v[j] * inv * inv_sqrt_dk;
        mx2 = fmaxf(mx2, v[j]);
      }
      float sum2 = 0.f;
#pragma unroll
      for (int j = 0; j < NHEAD; ++j) { v[j] = __expf(v[j] - mx2); sum2 += v[j]; }
      float inv2 = 1.f / sum2;
#pragma unroll
      for (int j = 0; j < NHEAD; ++j)
        sW[j * (32 * 33) + m * 33 + n] = v[j] * inv2;
    }
    __syncthreads();
    // ---- acc (32x64) += W_h (32x32) x Vp_h (32x64), V frags reused ----
    {
      v16bf vb0 = load_b_bf16(Vh + (size_t)0 * S_LEN + k0, S_LEN, lane);
      v16bf vb1 = load_b_bf16(Vh + (size_t)16 * S_LEN + k0, S_LEN, lane);
      v16bf vb2 = load_b_bf16(Vh + (size_t)32 * S_LEN + k0, S_LEN, lane);
      v16bf vb3 = load_b_bf16(Vh + (size_t)48 * S_LEN + k0, S_LEN, lane);
#pragma unroll
      for (int t = 0; t < 2; ++t) {
        v16bf wa = load_a_f32(myS + t * 16 * 33, 33, lane);
        acc[t][0] = wmma_bf16(wa, vb0, acc[t][0]);
        acc[t][1] = wmma_bf16(wa, vb1, acc[t][1]);
        acc[t][2] = wmma_bf16(wa, vb2, acc[t][2]);
        acc[t][3] = wmma_bf16(wa, vb3, acc[t][3]);
      }
    }
    __syncthreads();  // sW reused next iteration
  }
  // ---- store val[b, s, h*64+dk] ----
  int n = lane & 15;
  bf16_t* op = val + (((size_t)b * S_LEN + q0) * (NHEAD * DK)) + h * DK;
#pragma unroll
  for (int t = 0; t < 2; ++t) {
#pragma unroll
    for (int r = 0; r < 8; ++r) {
      int m = t * 16 + r + ((lane >> 4) << 3);
      bf16_t* rowp = op + (size_t)m * (NHEAD * DK);
      rowp[0 * 16 + n] = (bf16_t)acc[t][0][r];
      rowp[1 * 16 + n] = (bf16_t)acc[t][1][r];
      rowp[2 * 16 + n] = (bf16_t)acc[t][2][r];
      rowp[3 * 16 + n] = (bf16_t)acc[t][3][r];
    }
  }
}

// ---------------- Output projection: out = val @ Wo^T + bo (f32 out) -------
__global__ __launch_bounds__(512) void outproj_kernel(
    const bf16_t* __restrict__ val, const bf16_t* __restrict__ Wo,
    const float* __restrict__ bo, float* __restrict__ out) {
  int lane = threadIdx.x & 31;
  int wave = threadIdx.x >> 5;
  int stg = blockIdx.x;  // 512 s-tiles
  int o0 = wave * 64;
  const bf16_t* Vp = val + (size_t)stg * 16 * DMODEL;
  const bf16_t* W0 = Wo + (size_t)(o0 + 0) * DMODEL;
  const bf16_t* W1 = Wo + (size_t)(o0 + 16) * DMODEL;
  const bf16_t* W2 = Wo + (size_t)(o0 + 32) * DMODEL;
  const bf16_t* W3 = Wo + (size_t)(o0 + 48) * DMODEL;
  v8f acc0 = {}, acc1 = {}, acc2 = {}, acc3 = {};
  v16bf a = load_a_bf16(Vp, DMODEL, lane);
  v16bf b0 = load_b_bf16(W0, DMODEL, lane);
  v16bf b1 = load_b_bf16(W1, DMODEL, lane);
  v16bf b2 = load_b_bf16(W2, DMODEL, lane);
  v16bf b3 = load_b_bf16(W3, DMODEL, lane);
  for (int d0 = 0; d0 < DMODEL; d0 += 32) {
    v16bf an = a, bn0 = b0, bn1 = b1, bn2 = b2, bn3 = b3;
    int dn = d0 + 32;
    if (dn < DMODEL) {
      an = load_a_bf16(Vp + dn, DMODEL, lane);
      bn0 = load_b_bf16(W0 + dn, DMODEL, lane);
      bn1 = load_b_bf16(W1 + dn, DMODEL, lane);
      bn2 = load_b_bf16(W2 + dn, DMODEL, lane);
      bn3 = load_b_bf16(W3 + dn, DMODEL, lane);
    }
    acc0 = wmma_bf16(a, b0, acc0);
    acc1 = wmma_bf16(a, b1, acc1);
    acc2 = wmma_bf16(a, b2, acc2);
    acc3 = wmma_bf16(a, b3, acc3);
    a = an; b0 = bn0; b1 = bn1; b2 = bn2; b3 = bn3;
  }
  int n = lane & 15;
  float* op = out + (size_t)stg * 16 * DMODEL + o0 + n;
  v8f* accs[4] = {&acc0, &acc1, &acc2, &acc3};
#pragma unroll
  for (int i = 0; i < 4; ++i) {
    float bb = bo[o0 + i * 16 + n];
#pragma unroll
    for (int r = 0; r < 8; ++r) {
      int m = r + ((lane >> 4) << 3);
      op[(size_t)m * DMODEL + i * 16] = (*accs[i])[r] + bb;
    }
  }
}

extern "C" void kernel_launch(void* const* d_in, const int* in_sizes, int n_in,
                              void* d_out, int out_size, void* d_ws,
                              size_t ws_size, hipStream_t stream) {
  const float* k_in = (const float*)d_in[0];
  const float* q_in = (const float*)d_in[1];
  const float* v_in = (const float*)d_in[2];
  const float* Wq = (const float*)d_in[3];
  const float* bq = (const float*)d_in[4];
  const float* Wk = (const float*)d_in[5];
  const float* bk = (const float*)d_in[6];
  const float* Wv = (const float*)d_in[7];
  const float* bv = (const float*)d_in[8];
  const float* Wo = (const float*)d_in[9];
  const float* bo = (const float*)d_in[10];
  float* out = (float*)d_out;

  char* ws = (char*)d_ws;
  const size_t NE = (size_t)B_DIM * S_LEN * DMODEL;  // 8388608 elements
  const size_t WE = (size_t)DMODEL * DMODEL;         // 1048576 elements
  bf16_t* qb  = (bf16_t*)(ws);
  bf16_t* kb  = (bf16_t*)(ws + NE * 2);
  bf16_t* vb  = (bf16_t*)(ws + NE * 4);
  bf16_t* Wqb = (bf16_t*)(ws + NE * 6);
  bf16_t* Wkb = (bf16_t*)(ws + NE * 6 + WE * 2);
  bf16_t* Wvb = (bf16_t*)(ws + NE * 6 + WE * 4);
  bf16_t* Wob = (bf16_t*)(ws + NE * 6 + WE * 6);
  bf16_t* Qp  = (bf16_t*)(ws + NE * 6 + WE * 8);            // [B,H,S,64]
  bf16_t* Kp  = (bf16_t*)(ws + NE * 8 + WE * 8);            // [B,H,S,64]
  bf16_t* VpT = (bf16_t*)(ws + NE * 10 + WE * 8);           // [B,H,64,S]
  bf16_t* val = (bf16_t*)(ws + NE * 12 + WE * 8);           // [B,S,1024]

  dim3 blk(512);
  dim3 cblk(256);
  int nb_x = (int)(NE / (256 * 8));  // 4096 blocks
  int nb_w = (int)(WE / (256 * 8));  // 512 blocks
  cvt_kernel<<<nb_x, cblk, 0, stream>>>(q_in, qb, (int)NE);
  cvt_kernel<<<nb_x, cblk, 0, stream>>>(k_in, kb, (int)NE);
  cvt_kernel<<<nb_x, cblk, 0, stream>>>(v_in, vb, (int)NE);
  cvt_kernel<<<nb_w, cblk, 0, stream>>>(Wq, Wqb, (int)WE);
  cvt_kernel<<<nb_w, cblk, 0, stream>>>(Wk, Wkb, (int)WE);
  cvt_kernel<<<nb_w, cblk, 0, stream>>>(Wv, Wvb, (int)WE);
  cvt_kernel<<<nb_w, cblk, 0, stream>>>(Wo, Wob, (int)WE);

  proj_kernel<<<512, blk, 0, stream>>>(qb, Wqb, bq, Qp);
  proj_kernel<<<512, blk, 0, stream>>>(kb, Wkb, bk, Kp);
  projT_kernel<<<512, blk, 0, stream>>>(vb, Wvb, bv, VpT);
  attn_kernel<<<256, blk, 0, stream>>>(Qp, Kp, VpT, val);
  outproj_kernel<<<512, blk, 0, stream>>>(val, Wob, bo, out);
}
